// VanillaRNN_55130200211525
// MI455X (gfx1250) — matrix operand order repaired and checked by reading, not compiled, not run on hardware
//
#include <hip/hip_runtime.h>
#include <hip/hip_bf16.h>

// ---------------------------------------------------------------------------
// Vanilla RNN: B=512, T=256, S=64, I=32, H=1024
//   per step: h1 = sig([x|u] W1^T + b1); h2 = sig(h1 W2^T + b2); x = h2 W3^T + b3
//
// MI455X (gfx1250) strategy:
//  - Batch-partitioned persistent blocks: 32 blocks x 16 batch rows, 8 wave32s
//    (256 waves total) -> max sync-free parallelism; each block runs all T
//    steps for its slice, no grid-wide sync ever.
//  - Activations LDS-resident: aBuf(16x96) + h1(16x1024) + h2(16x1024) bf16
//    = ~67 KB of the 320 KB/WGP LDS.
//  - Weights converted once to bf16 in d_ws (~2.3 MB), streamed from L2
//    (192 MB L2 holds all weights; HBM traffic ~= u-in (16MB) + y-out (32MB)).
//  - Math: v_wmma_f32_16x16x32_bf16, f32 accumulate, f32 bias+sigmoid.
//  - Sigmoid uses v_exp_f32 + v_rcp_f32 (no IEEE div expansion); biases are
//    hoisted to registers outside the T loop.
//  - Layer-2 inner loop: one A-fragment (software-pipelined) feeds 8 WMMAs
//    into 8 live accumulator tiles; W2 read exactly once per block per step.
//  - Layer-3: 4 waves, 2 interleaved accumulators each to break the WMMA RAW
//    chain.
// ---------------------------------------------------------------------------

typedef __attribute__((ext_vector_type(16))) __bf16 v16bf;
typedef __attribute__((ext_vector_type(8)))  __bf16 v8bf;
typedef __attribute__((ext_vector_type(8)))  float  v8f;

#define B_  512
#define T_  256
#define S_  64
#define I_  32
#define H_  1024
#define BT  16      // batch tile per block (one 16-row m-tile)
#define NTHREADS 256

static __device__ __forceinline__ v16bf mk16(v8bf lo, v8bf hi) {
  union { v16bf w; v8bf h[2]; } u;
  u.h[0] = lo; u.h[1] = hi;
  return u.w;
}

static __device__ __forceinline__ v8f wmma_bf16(v16bf a, v16bf b, v8f c) {
  // D = A(16x32) * B(32x16) + C, f32 accumulate
  return __builtin_amdgcn_wmma_f32_16x16x32_bf16(
      /*neg_a=*/false, a, /*neg_b=*/false, b,
      /*c_mod=*/(short)0, c, /*reuse_a=*/false, /*reuse_b=*/false);
}

static __device__ __forceinline__ float sigmoidf_(float x) {
  // v_exp_f32 + v_rcp_f32: avoid the IEEE-exact div expansion
  return __builtin_amdgcn_rcpf(1.0f + __expf(-x));
}

// f32 -> bf16 weight conversion (one-time per launch, deterministic)
__global__ void cvt_f32_bf16(const float* __restrict__ src,
                             __bf16* __restrict__ dst, int n) {
  int i = blockIdx.x * blockDim.x + threadIdx.x;
  if (i < n) dst[i] = (__bf16)src[i];
}

__global__ __launch_bounds__(NTHREADS)
void rnn_persistent(const float* __restrict__ x0,   // (B,1,S)
                    const float* __restrict__ u,    // (B,T,I)
                    const float* __restrict__ b1,   // (H)
                    const float* __restrict__ b2,   // (H)
                    const float* __restrict__ b3,   // (S)
                    const __bf16* __restrict__ W1b, // (H,96)  row-major
                    const __bf16* __restrict__ W2b, // (H,H)
                    const __bf16* __restrict__ W3b, // (S,H)
                    float* __restrict__ out)        // (B,T,S)
{
  __shared__ __align__(16) __bf16 aBuf[BT * 96];   // [x(0:64) | u(64:96)]
  __shared__ __align__(16) __bf16 hA[BT * H_];     // h1
  __shared__ __align__(16) __bf16 hB[BT * H_];     // h2

  const int tid   = threadIdx.x;
  const int wave  = tid >> 5;    // 0..7
  const int lane  = tid & 31;
  const int lhalf = lane >> 4;   // 0/1 : which half-wave
  const int lmod  = lane & 15;   // column within tile (B/C/D) or row (A)
  const int b0    = blockIdx.x * BT;

  const int arow  = lmod;        // A-matrix row for this lane (single m-tile)
  const int mbase = lhalf * 8;   // C/D row base for this lane

  // ---- hoist step-invariant per-lane biases into registers ----
  float bias1[8], bias2[8];
  #pragma unroll
  for (int nt = 0; nt < 8; ++nt) {
    bias1[nt] = b1[(wave * 8 + nt) * 16 + lmod];
    bias2[nt] = b2[wave * 128 + nt * 16 + lmod];
  }
  const float bias3 = (wave < 4) ? b3[wave * 16 + lmod] : 0.0f;

  // ---- init: x part of aBuf from x0 (f32 -> bf16); 16 rows x 16 float4 ----
  {
    const int r = tid >> 4, c4 = tid & 15;
    const float4 v = *(const float4*)&x0[(b0 + r) * S_ + c4 * 4];
    __bf16* p = &aBuf[r * 96 + c4 * 4];
    p[0] = (__bf16)v.x; p[1] = (__bf16)v.y;
    p[2] = (__bf16)v.z; p[3] = (__bf16)v.w;
  }

  for (int t = 0; t < T_; ++t) {
    // ---- stage u_t into aBuf[:, 64:96]; 16 rows x 8 float4 = 128 lanes ----
    if (tid < BT * 8) {
      const int r = tid >> 3, c4 = tid & 7;
      const float4 v = *(const float4*)&u[((size_t)(b0 + r) * T_ + t) * I_ + c4 * 4];
      __bf16* p = &aBuf[r * 96 + 64 + c4 * 4];
      p[0] = (__bf16)v.x; p[1] = (__bf16)v.y;
      p[2] = (__bf16)v.z; p[3] = (__bf16)v.w;
    }
    __syncthreads();

    // ---- layer 1: h1 = sigmoid(aBuf(16x96) @ W1^T + b1) -> hA ----
    // wave w owns n-tiles w*8 .. w*8+7
    {
      for (int nt = 0; nt < 8; ++nt) {
        const int ncol = (wave * 8 + nt) * 16 + lmod;  // output column
        v8f acc = {};
        #pragma unroll
        for (int kf = 0; kf < 3; ++kf) {
          const int k0 = kf * 32;
          v8bf alo = *(const v8bf*)&aBuf[arow * 96 + k0 + lhalf * 8];
          v8bf ahi = *(const v8bf*)&aBuf[arow * 96 + k0 + 16 + lhalf * 8];
          const __bf16* bp = &W1b[ncol * 96 + k0];
          v8bf blo = *(const v8bf*)(bp + lhalf * 8);
          v8bf bhi = *(const v8bf*)(bp + 16 + lhalf * 8);
          acc = wmma_bf16(mk16(alo, ahi), mk16(blo, bhi), acc);
        }
        #pragma unroll
        for (int v = 0; v < 8; ++v)
          hA[(mbase + v) * H_ + ncol] = (__bf16)sigmoidf_(acc[v] + bias1[nt]);
      }
    }
    __syncthreads();

    // ---- layer 2: h2 = sigmoid(hA(16x1024) @ W2^T + b2) -> hB ----
    // wave w owns n-tiles w*8 .. w*8+7; 8 live accumulators; A pipelined
    {
      const int nbase = wave * 128;                    // first output column
      v8f acc[8] = {};
      v8bf alo = *(const v8bf*)&hA[arow * H_ + lhalf * 8];
      v8bf ahi = *(const v8bf*)&hA[arow * H_ + 16 + lhalf * 8];
      for (int kf = 0; kf < 32; ++kf) {
        const int k0 = kf * 32;
        const v16bf a = mk16(alo, ahi);
        if (kf < 31) {                                 // pipeline next A-frag
          alo = *(const v8bf*)&hA[arow * H_ + k0 + 32 + lhalf * 8];
          ahi = *(const v8bf*)&hA[arow * H_ + k0 + 48 + lhalf * 8];
          // pull next W2 K-panel toward the WGP (global_prefetch)
          __builtin_prefetch(&W2b[(nbase + lmod) * H_ + k0 + 32 + lhalf * 16], 0, 1);
        }
        #pragma unroll
        for (int nt = 0; nt < 8; ++nt) {
          const int ncol = nbase + nt * 16 + lmod;
          const __bf16* bp = &W2b[ncol * H_ + k0];
          v8bf blo = *(const v8bf*)(bp + lhalf * 8);
          v8bf bhi = *(const v8bf*)(bp + 16 + lhalf * 8);
          acc[nt] = wmma_bf16(a, mk16(blo, bhi), acc[nt]);
        }
      }
      #pragma unroll
      for (int nt = 0; nt < 8; ++nt) {
        const int ncol = nbase + nt * 16 + lmod;
        #pragma unroll
        for (int v = 0; v < 8; ++v)
          hB[(mbase + v) * H_ + ncol] = (__bf16)sigmoidf_(acc[nt][v] + bias2[nt]);
      }
    }
    __syncthreads();

    // ---- layer 3: x_new = hB(16x1024) @ W3^T + b3 -> out + aBuf[:,0:64] ----
    // 4 n-tiles (S=64): waves 0..3, two interleaved accumulators each
    if (wave < 4) {
      const int ncol = wave * 16 + lmod;
      v8f acc0 = {}, acc1 = {};
      for (int kf = 0; kf < 32; kf += 2) {
        const int k0 = kf * 32;
        {
          v8bf alo = *(const v8bf*)&hB[arow * H_ + k0 + lhalf * 8];
          v8bf ahi = *(const v8bf*)&hB[arow * H_ + k0 + 16 + lhalf * 8];
          const __bf16* bp = &W3b[ncol * H_ + k0];
          v8bf blo = *(const v8bf*)(bp + lhalf * 8);
          v8bf bhi = *(const v8bf*)(bp + 16 + lhalf * 8);
          acc0 = wmma_bf16(mk16(alo, ahi), mk16(blo, bhi), acc0);
        }
        {
          const int k1 = k0 + 32;
          v8bf alo = *(const v8bf*)&hB[arow * H_ + k1 + lhalf * 8];
          v8bf ahi = *(const v8bf*)&hB[arow * H_ + k1 + 16 + lhalf * 8];
          const __bf16* bp = &W3b[ncol * H_ + k1];
          v8bf blo = *(const v8bf*)(bp + lhalf * 8);
          v8bf bhi = *(const v8bf*)(bp + 16 + lhalf * 8);
          acc1 = wmma_bf16(mk16(alo, ahi), mk16(blo, bhi), acc1);
        }
      }
      const v8f acc = acc0 + acc1;
      #pragma unroll
      for (int v = 0; v < 8; ++v) {
        const float xv = acc[v] + bias3;
        out[((size_t)(b0 + mbase + v) * T_ + t) * S_ + ncol] = xv;   // f32 out
        aBuf[(mbase + v) * 96 + ncol] = (__bf16)xv;                  // next state
      }
    }
    __syncthreads();
  }
}

extern "C" void kernel_launch(void* const* d_in, const int* in_sizes, int n_in,
                              void* d_out, int out_size, void* d_ws, size_t ws_size,
                              hipStream_t stream) {
  (void)in_sizes; (void)n_in; (void)out_size; (void)ws_size;
  const float* x0 = (const float*)d_in[0];
  const float* u  = (const float*)d_in[1];
  const float* W1 = (const float*)d_in[2];
  const float* b1 = (const float*)d_in[3];
  const float* W2 = (const float*)d_in[4];
  const float* b2 = (const float*)d_in[5];
  const float* W3 = (const float*)d_in[6];
  const float* b3 = (const float*)d_in[7];
  float* out = (float*)d_out;

  // bf16 weight staging in workspace (needs ~2.32 MB of d_ws)
  __bf16* W1b = (__bf16*)d_ws;                 // 1024*96
  __bf16* W2b = W1b + H_ * 96;                 // 1024*1024
  __bf16* W3b = W2b + H_ * H_;                 // 64*1024

  const int n1 = H_ * 96, n2 = H_ * H_, n3 = S_ * H_;
  cvt_f32_bf16<<<(n1 + 255) / 256, 256, 0, stream>>>(W1, W1b, n1);
  cvt_f32_bf16<<<(n2 + 255) / 256, 256, 0, stream>>>(W2, W2b, n2);
  cvt_f32_bf16<<<(n3 + 255) / 256, 256, 0, stream>>>(W3, W3b, n3);

  rnn_persistent<<<B_ / BT, NTHREADS, 0, stream>>>(
      x0, u, b1, b2, b3, W1b, W2b, W3b, out);
}
